// SA_Layer_20899310862674
// MI455X (gfx1250) — matrix operand order repaired and checked by reading, not compiled, not run on hardware
//
#include <hip/hip_runtime.h>
#include <hip/hip_bf16.h>

typedef __attribute__((ext_vector_type(16))) _Float16 v16h;
typedef __attribute__((ext_vector_type(8)))  float    v8f;

#define BB 16
#define NN 2048
#define DD 128
#define DQ 32
#define BN_EPS 1e-3f

static __device__ __forceinline__ v8f wmma_f16(v16h a, v16h b, v8f c) {
  // D = A(16x32 f16) * B(32x16 f16) + C(16x16 f32)
  return __builtin_amdgcn_wmma_f32_16x16x32_f16(false, a, false, b, (short)0, c,
                                                false, false);
}

// LDS byte offset of a __shared__ object (generic addr low 32 bits = LDS offset)
static __device__ __forceinline__ uint32_t lds_offset(const void* p) {
  return (uint32_t)(uintptr_t)p;
}

// ---------------------------------------------------------------------------
// K1: q = BN(x@Wq) -> f16 [B,N,32]; v = BN(x@Wv) -> f16 [B,N,128];
//     mask_offset = -1000*mask+1; Woff -> f16 copy.
// ---------------------------------------------------------------------------
__global__ void k1_prep(const float* __restrict__ x, const float* __restrict__ mask,
                        const float* __restrict__ Wq, const float* __restrict__ gq,
                        const float* __restrict__ bq, const float* __restrict__ mq,
                        const float* __restrict__ vq,
                        const float* __restrict__ Wv, const float* __restrict__ gv,
                        const float* __restrict__ bv, const float* __restrict__ mv,
                        const float* __restrict__ vv,
                        const float* __restrict__ Woff,
                        _Float16* __restrict__ qh, _Float16* __restrict__ vh,
                        _Float16* __restrict__ woffh, float* __restrict__ mko)
{
  __shared__ float xs[DD];
  const int row = blockIdx.x;     // 0..B*N-1
  const int t   = threadIdx.x;    // 0..127
  xs[t] = x[(size_t)row * DD + t];
  __syncthreads();

  float s = 0.f;
  #pragma unroll 4
  for (int d = 0; d < DD; ++d) s = fmaf(xs[d], Wv[d * DD + t], s);
  float sc = gv[t] * rsqrtf(vv[t] + BN_EPS);
  vh[(size_t)row * DD + t] = (_Float16)((s - mv[t]) * sc + bv[t]);

  if (t < DQ) {
    float sq = 0.f;
    #pragma unroll 4
    for (int d = 0; d < DD; ++d) sq = fmaf(xs[d], Wq[d * DQ + t], sq);
    float scq = gq[t] * rsqrtf(vq[t] + BN_EPS);
    qh[(size_t)row * DQ + t] = (_Float16)((sq - mq[t]) * scq + bq[t]);
  }
  if (t == 0) mko[row] = fmaf(-1000.f, mask[row], 1.f);
  if (blockIdx.x == 0)
    for (int i = t; i < DD * DD; i += blockDim.x) woffh[i] = (_Float16)Woff[i];
}

__global__ void k_zero(float* p, int n) {
  int i = blockIdx.x * blockDim.x + threadIdx.x;
  if (i < n) p[i] = 0.f;
}

// ---------------------------------------------------------------------------
// K2: per (b, 16-row stripe): energy via WMMA (K=32 -> one wmma per 16x16
// tile), + mask matrix, row softmax, zero-mask, column-sum atomics.
// Whole 16x2048 f32 stripe kept in dynamic LDS (128KB; CDNA5 WGP has 320KB).
// Software-pipelined: tile j's WMMA overlaps tile j-1's epilogue, so the
// WMMA->VALU RAW hazard co-exec slots do useful work instead of v_nops.
// ---------------------------------------------------------------------------
__global__ void k2_energy_softmax(const _Float16* __restrict__ qh,
                                  const float* __restrict__ mko,
                                  float* __restrict__ attOut,
                                  float* __restrict__ colsum)
{
  extern __shared__ float smem[];
  float* stripe = smem;               // [16][2048]
  float* redmax = stripe + 16 * NN;   // [4][16]
  float* rowmax = redmax + 64;        // [16]
  float* rowsum = rowmax + 16;        // [16]
  float* mrow   = rowsum + 16;        // [16]

  const int b    = blockIdx.x >> 7;           // N/16 = 128 tiles per batch
  const int row0 = (blockIdx.x & 127) * 16;
  const int tid  = threadIdx.x;
  const int wv   = tid >> 5;
  const int lane = tid & 31;
  const int m    = lane & 15;
  const int hi   = lane >> 4;

  if (tid < 16) { mrow[tid] = mko[b * NN + row0 + tid]; rowsum[tid] = 0.f; }
  __syncthreads();

  // A operand: q rows row0..row0+15 (lane m holds contiguous K run
  // kb..kb+15, kb = hi*16, matching B)
  const _Float16* qa = qh + ((size_t)b * NN + row0 + m) * DQ + hi * 16;
  v16h A;
  #pragma unroll
  for (int i = 0; i < 8; ++i) { A[i] = qa[i]; A[i + 8] = qa[8 + i]; }

  float mr[8];
  #pragma unroll
  for (int r = 0; r < 8; ++r) mr[r] = mrow[r + hi * 8];

  float rmax[8];
  #pragma unroll
  for (int r = 0; r < 8; ++r) rmax[r] = -3.4e38f;

  v8f Cp;
  int colp = -1;
  for (int jt = wv; jt < NN / 16; jt += 4) {
    const int col0 = jt * 16;
    // B operand: B[k,n] = q[col0+n][k]; lane n holds contiguous K run of 16
    const _Float16* qb = qh + ((size_t)b * NN + col0 + m) * DQ + hi * 16;
    v16h Bv = *(const v16h*)qb;             // 32B aligned vector load
    v8f C = {};
    C = wmma_f16(A, Bv, C);                 // in flight while we retire colp
    if (colp >= 0) {
      const float mc = mko[b * NN + colp + m];
      #pragma unroll
      for (int r = 0; r < 8; ++r) {         // C layout: M = r + 8*hi, N = m
        const int Mr = r + hi * 8;
        float e = Cp[r] + mr[r] * mc - 1.0f;
        stripe[Mr * NN + colp + m] = e;
        rmax[r] = fmaxf(rmax[r], e);
      }
    }
    Cp = C;
    colp = col0;
  }
  if (colp >= 0) {
    const float mc = mko[b * NN + colp + m];
    #pragma unroll
    for (int r = 0; r < 8; ++r) {
      const int Mr = r + hi * 8;
      float e = Cp[r] + mr[r] * mc - 1.0f;
      stripe[Mr * NN + colp + m] = e;
      rmax[r] = fmaxf(rmax[r], e);
    }
  }

  // row max: reduce across the 16 lanes of each half-wave, then across waves
  #pragma unroll
  for (int msk = 1; msk < 16; msk <<= 1)
    #pragma unroll
    for (int r = 0; r < 8; ++r)
      rmax[r] = fmaxf(rmax[r], __shfl_xor(rmax[r], msk, 32));
  if (m == 0)
    #pragma unroll
    for (int r = 0; r < 8; ++r) redmax[wv * 16 + r + hi * 8] = rmax[r];
  __syncthreads();
  if (tid < 16) {
    float v0 = redmax[tid];
    for (int w = 1; w < 4; ++w) v0 = fmaxf(v0, redmax[w * 16 + tid]);
    rowmax[tid] = v0;
  }
  __syncthreads();

  // pass 2: exp + row sums (softmax denominator over ALL columns)
  for (int rr = 0; rr < 16; ++rr) {
    const float rm = rowmax[rr];
    float s = 0.f;
    for (int cc = tid; cc < NN; cc += blockDim.x) {
      float ex = __expf(stripe[rr * NN + cc] - rm);
      stripe[rr * NN + cc] = ex;
      s += ex;
    }
    atomicAdd(&rowsum[rr], s);
  }
  __syncthreads();

  // pass 3: att_z = softmax * (mask_matrix==0); accumulate column sums
  for (int cc = tid; cc < NN; cc += blockDim.x) {
    const float mc = mko[b * NN + cc];
    float cs = 0.f;
    for (int rr = 0; rr < 16; ++rr) {
      float mm = mrow[rr] * mc - 1.0f;
      float az = (mm == 0.0f) ? (stripe[rr * NN + cc] / rowsum[rr]) : 0.f;
      stripe[rr * NN + cc] = az;
      cs += az;
    }
    atomicAdd(&colsum[b * NN + cc], cs);
  }
  __syncthreads();

  // coalesced store of the stripe
  for (int idx = tid; idx < 16 * NN; idx += blockDim.x) {
    const int rr = idx >> 11, cc = idx & (NN - 1);
    attOut[((size_t)b * NN + row0 + rr) * NN + cc] = stripe[idx];
  }
}

// ---------------------------------------------------------------------------
// K3: per (b, 16-column tile m0): normalize att by colsum (rewrite final
// attention in place), self_att = att_norm^T @ v via WMMA, then fused
// off = relu(BN((x - self_att) @ Woff)), out = x + off  (more WMMA).
// Woff is staged to LDS by the Tensor Data Mover (TENSORcnt); v chunks are
// staged with GLOBAL_LOAD_ASYNC_TO_LDS_B128 (ASYNCcnt) to overlap with WMMA.
// ---------------------------------------------------------------------------
__global__ void k3_selfatt_out(const float* __restrict__ x,
                               const _Float16* __restrict__ vh,
                               const _Float16* __restrict__ woffh,
                               const float* __restrict__ colsum,
                               const float* __restrict__ goff,
                               const float* __restrict__ boff,
                               const float* __restrict__ meano,
                               const float* __restrict__ varo,
                               float* __restrict__ attOut,
                               float* __restrict__ out)
{
  __shared__ __align__(16) _Float16 attH[32 * 16]; // att tile, normalized
  __shared__ __align__(16) _Float16 vLds[32 * DD]; // v chunk
  __shared__ __align__(16) _Float16 wLds[DD * DD]; // Woff (f16), TDM-resident
  __shared__ __align__(16) _Float16 yH[16 * DD];   // y = x - self_att
  __shared__ float invc[16];

  const int b   = blockIdx.x >> 7;
  const int m0  = (blockIdx.x & 127) * 16;
  const int tid = threadIdx.x, wv = tid >> 5, lane = tid & 31;
  const int ml  = lane & 15, hi = lane >> 4;

  if (tid < 16) invc[tid] = 1.0f / (1e-9f + colsum[b * NN + m0 + tid]);

#if __has_builtin(__builtin_amdgcn_tensor_load_to_lds)
  // TDM: one descriptor moves the whole 16384-element (32KB) f16 Woff into LDS.
  // amdgpu-toolchain (clang-23) signature: (u32x4, i32x8, i32x4, i32x4, i32x8, i32)
  typedef unsigned int u32x4 __attribute__((ext_vector_type(4)));
  typedef int          i32x8 __attribute__((ext_vector_type(8)));
  typedef int          i32x4 __attribute__((ext_vector_type(4)));
  if (wv == 0) {
    const uint64_t ga  = (uint64_t)(uintptr_t)woffh;
    const uint32_t lds = lds_offset(wLds);
    u32x4 g0;
    g0[0] = 1u;                                        // count=1, user mode
    g0[1] = lds;                                       // lds_addr (bytes)
    g0[2] = (uint32_t)ga;                              // global_addr[31:0]
    g0[3] = (uint32_t)((ga >> 32) & 0x1FFFFFFu) | (2u << 30); // [56:32], type=2
    i32x8 g1;
    g1[0] = 1 << 16;                                   // data_size = 2 bytes
    g1[1] = (int)(16384u << 16);                       // tensor_dim0 = 16384
    g1[2] = 0;                                         // dim0 hi / tensor_dim1
    g1[3] = (int)(16384u << 16);                       // tile_dim0 = 16384
    g1[4] = 0;                                         // tile_dim1/2 unused (1D)
    g1[5] = 16384;                                     // tensor_dim0_stride
    g1[6] = 0;
    g1[7] = 0;
    i32x4 z4 = {0, 0, 0, 0};
    i32x8 z8 = {0, 0, 0, 0, 0, 0, 0, 0};
    __builtin_amdgcn_tensor_load_to_lds(g0, g1, z4, z4, z8, 0);
  }
  __builtin_amdgcn_s_wait_tensorcnt(0);
#else
  for (int i = tid; i < DD * DD / 8; i += blockDim.x)
    ((uint4*)wLds)[i] = ((const uint4*)woffh)[i];
#endif
  __syncthreads();

  v8f acc0 = {}, acc1 = {};
  const int d0a = (2 * wv) * 16, d0b = (2 * wv + 1) * 16;

  for (int n0 = 0; n0 < NN; n0 += 32) {
    // async-copy v chunk [32 x 128] f16 -> LDS (8KB = 512 x b128, 4 per lane)
    {
      const char* gsrc = (const char*)(vh + ((size_t)b * NN + n0) * DD);
      for (int idx = tid; idx < (32 * DD * 2) / 16; idx += blockDim.x) {
        uint32_t lofs = lds_offset((const char*)vLds + idx * 16);
        uint64_t gaddr = (uint64_t)(uintptr_t)(gsrc + idx * 16);
        asm volatile("global_load_async_to_lds_b128 %0, %1, off"
                     :: "v"(lofs), "v"(gaddr) : "memory");
      }
    }
    // meanwhile: load att_z tile through VGPRs, normalize, rewrite final att
    for (int idx = tid; idx < 512; idx += blockDim.x) {
      const int i = idx >> 4, j = idx & 15;
      const size_t g = ((size_t)b * NN + n0 + i) * NN + m0 + j;
      float fin = attOut[g] * invc[j];
      attOut[g] = fin;
      attH[i * 16 + j] = (_Float16)fin;
    }
    asm volatile("s_wait_asynccnt 0x0" ::: "memory");
    __syncthreads();

    v16h A;                               // A[m,k] = att_norm[n0+k, m0+m]
    #pragma unroll
    for (int i = 0; i < 8; ++i) {
      A[i]     = attH[(hi * 16 + i) * 16 + ml];
      A[i + 8] = attH[(hi * 16 + 8 + i) * 16 + ml];
    }
    v16h B0, B1;                          // B[k,n] = v[n0+k][d0+n]
    #pragma unroll
    for (int i = 0; i < 16; ++i) {
      B0[i] = vLds[(hi * 16 + i) * DD + d0a + ml];
      B1[i] = vLds[(hi * 16 + i) * DD + d0b + ml];
    }
    acc0 = wmma_f16(A, B0, acc0);
    acc1 = wmma_f16(A, B1, acc1);
    __syncthreads();
  }

  // y = x - self_att (f16 staging for next GEMM)
  #pragma unroll
  for (int r = 0; r < 8; ++r) {
    const int Mr = r + hi * 8;
    const size_t xrow = ((size_t)b * NN + m0 + Mr) * DD;
    yH[Mr * DD + d0a + ml] = (_Float16)(x[xrow + d0a + ml] - acc0[r]);
    yH[Mr * DD + d0b + ml] = (_Float16)(x[xrow + d0b + ml] - acc1[r]);
  }
  __syncthreads();

  // off = relu(BN(y @ Woff)); out = x + off
  v8f o0 = {}, o1 = {};
  for (int kc = 0; kc < 4; ++kc) {
    v16h A2;
    #pragma unroll
    for (int i = 0; i < 8; ++i) {
      A2[i]     = yH[ml * DD + kc * 32 + hi * 16 + i];
      A2[i + 8] = yH[ml * DD + kc * 32 + hi * 16 + 8 + i];
    }
    v16h W0, W1;
    #pragma unroll
    for (int i = 0; i < 16; ++i) {
      W0[i] = wLds[(kc * 32 + hi * 16 + i) * DD + d0a + ml];
      W1[i] = wLds[(kc * 32 + hi * 16 + i) * DD + d0b + ml];
    }
    o0 = wmma_f16(A2, W0, o0);
    o1 = wmma_f16(A2, W1, o1);
  }
  #pragma unroll
  for (int r = 0; r < 8; ++r) {
    const int Mr = r + hi * 8;
    const size_t xrow = ((size_t)b * NN + m0 + Mr) * DD;
    {
      const int ch = d0a + ml;
      float sc  = goff[ch] * rsqrtf(varo[ch] + BN_EPS);
      float off = fmaxf(0.f, (o0[r] - meano[ch]) * sc + boff[ch]);
      out[xrow + ch] = x[xrow + ch] + off;
    }
    {
      const int ch = d0b + ml;
      float sc  = goff[ch] * rsqrtf(varo[ch] + BN_EPS);
      float off = fmaxf(0.f, (o1[r] - meano[ch]) * sc + boff[ch]);
      out[xrow + ch] = x[xrow + ch] + off;
    }
  }
}

// ---------------------------------------------------------------------------
extern "C" void kernel_launch(void* const* d_in, const int* in_sizes, int n_in,
                              void* d_out, int out_size, void* d_ws, size_t ws_size,
                              hipStream_t stream) {
  (void)in_sizes; (void)n_in; (void)out_size; (void)ws_size;
  const float* x     = (const float*)d_in[0];
  const float* mask  = (const float*)d_in[1];
  const float* Wq    = (const float*)d_in[2];
  const float* gq    = (const float*)d_in[3];
  const float* bq    = (const float*)d_in[4];
  const float* mq    = (const float*)d_in[5];
  const float* vq    = (const float*)d_in[6];
  const float* Wv    = (const float*)d_in[7];
  const float* gv    = (const float*)d_in[8];
  const float* bv    = (const float*)d_in[9];
  const float* mv    = (const float*)d_in[10];
  const float* vv    = (const float*)d_in[11];
  const float* Woff  = (const float*)d_in[12];
  const float* goff  = (const float*)d_in[13];
  const float* boff  = (const float*)d_in[14];
  const float* meano = (const float*)d_in[15];
  const float* varo  = (const float*)d_in[16];

  float* out    = (float*)d_out;                 // [B,N,D]
  float* attOut = out + (size_t)BB * NN * DD;    // [B,N,N]

  char* w = (char*)d_ws;
  _Float16* qh    = (_Float16*)w; w += (size_t)BB * NN * DQ * sizeof(_Float16);
  _Float16* vhp   = (_Float16*)w; w += (size_t)BB * NN * DD * sizeof(_Float16);
  _Float16* woffh = (_Float16*)w; w += (size_t)DD * DD * sizeof(_Float16);
  float*    mko   = (float*)w;    w += (size_t)BB * NN * sizeof(float);
  float*    colsum= (float*)w;    w += (size_t)BB * NN * sizeof(float);

  k1_prep<<<BB * NN, 128, 0, stream>>>(x, mask, Wq, gq, bq, mq, vq,
                                       Wv, gv, bv, mv, vv, Woff,
                                       qh, vhp, woffh, mko);
  k_zero<<<(BB * NN + 255) / 256, 256, 0, stream>>>(colsum, BB * NN);

  const int dynBytes = (16 * NN + 64 + 16 + 16 + 16) * (int)sizeof(float);
  (void)hipFuncSetAttribute((const void*)k2_energy_softmax,
                            hipFuncAttributeMaxDynamicSharedMemorySize, dynBytes);
  k2_energy_softmax<<<BB * (NN / 16), 128, dynBytes, stream>>>(qh, mko, attOut,
                                                               colsum);

  k3_selfatt_out<<<BB * (NN / 16), 128, 0, stream>>>(x, vhp, woffh, colsum,
                                                     goff, boff, meano, varo,
                                                     attOut, out);
}
